// BinaryFactoredLinear_8048768712883
// MI455X (gfx1250) — matrix operand order repaired and checked by reading, not compiled, not run on hardware
//
#include <hip/hip_runtime.h>

typedef __bf16 bf16_t;
typedef bf16_t v16bf __attribute__((ext_vector_type(16)));
typedef bf16_t v8bf  __attribute__((ext_vector_type(8)));
typedef float  v8f   __attribute__((ext_vector_type(8)));

// Problem dims (fixed by reference: B=4, S=2048, D_IN=D_OUT=4096, R=512)
#define MTOK 8192   // B*S tokens
#define DIN  4096
#define DOUT 4096
#define RR   512

__device__ __forceinline__ bf16_t sgnbf(float v) {
  float s = (v > 0.0f) ? 1.0f : ((v < 0.0f) ? -1.0f : 0.0f);
  return (bf16_t)s;
}

// xs[m, d] = bf16(x[m, d] * s2[d]);  8 elements per thread, contiguous along d.
__global__ void scale_x_kernel(const float* __restrict__ x,
                               const float* __restrict__ s2,
                               bf16_t* __restrict__ xs) {
  long long i = ((long long)blockIdx.x * blockDim.x + threadIdx.x) * 8;
  int d = (int)(i & (DIN - 1));
#pragma unroll
  for (int t = 0; t < 8; ++t)
    xs[i + t] = (bf16_t)(x[i + t] * s2[d + t]);
}

// usign[n, r] = sign(U_latent[n, r])  (same layout, elementwise)
__global__ void sign_u_kernel(const float* __restrict__ U,
                              bf16_t* __restrict__ usg) {
  long long i = ((long long)blockIdx.x * blockDim.x + threadIdx.x) * 8;
#pragma unroll
  for (int t = 0; t < 8; ++t)
    usg[i + t] = sgnbf(U[i + t]);
}

// vsT[r, d] = sign(V_latent[d, r]) : LDS-tiled transpose, both sides coalesced.
__global__ void sign_vT_kernel(const float* __restrict__ V,
                               bf16_t* __restrict__ vsT) {
  __shared__ bf16_t tile[32][33];
  const int rb = blockIdx.x * 32;  // r base
  const int db = blockIdx.y * 32;  // d base
  const int tx = threadIdx.x;      // 0..31
  const int ty = threadIdx.y;      // 0..7
#pragma unroll
  for (int yy = ty; yy < 32; yy += 8)
    tile[yy][tx] = sgnbf(V[(size_t)(db + yy) * RR + rb + tx]); // tile[d][r]
  __syncthreads();
#pragma unroll
  for (int yy = ty; yy < 32; yy += 8)
    vsT[(size_t)(rb + yy) * DIN + db + tx] = tile[tx][yy];
}

// C[M,N] = A[M,K] (bf16, row-major) x B[N,K]^T (bf16, row-major along K).
// Block: 8 waves, each wave owns a 64x64 tile -> block tile 128(M) x 256(N).
// OUT_F32=false : store bf16 (feeds GEMM2's A operand)
// OUT_F32=true  : store f32 with fused *s1 + bias epilogue
template <bool OUT_F32>
__global__ __launch_bounds__(256, 1)
void gemm_wmma_bf16(const bf16_t* __restrict__ A, const bf16_t* __restrict__ B,
                    void* __restrict__ Cout, int M, int N, int K,
                    const float* __restrict__ s1, const float* __restrict__ bias) {
  const int lane = threadIdx.x & 31;
  const int wave = threadIdx.x >> 5;   // 0..7
  const int wm   = wave >> 2;          // 0..1
  const int wn   = wave & 3;           // 0..3
  const int row  = lane & 15;          // matrix row (A) / column (B,D) index
  const int half = lane >> 4;          // K half-select

  const int M0 = blockIdx.x * 128 + wm * 64;
  const int N0 = blockIdx.y * 256 + wn * 64;

  v8f acc[4][4] = {};

  const bf16_t* Abase = A + (size_t)M0 * K;
  const bf16_t* Bbase = B + (size_t)N0 * K;

  for (int k0 = 0; k0 < K; k0 += 32) {
    v16bf a[4], b[4];
    // A fragment: K contiguous, lane half=0 -> K {0..15}, half=1 -> K {16..31}
#pragma unroll
    for (int i = 0; i < 4; ++i) {
      const bf16_t* p = Abase + (size_t)(16 * i + row) * K + k0 + half * 16;
      v8bf lo = *(const v8bf*)p;
      v8bf hi = *(const v8bf*)(p + 8);
      a[i] = __builtin_shufflevector(lo, hi, 0, 1, 2, 3, 4, 5, 6, 7,
                                     8, 9, 10, 11, 12, 13, 14, 15);
    }
    // B fragment: lane = column N0+16j+row, K contiguous (half*16 .. half*16+15)
#pragma unroll
    for (int j = 0; j < 4; ++j) {
      const bf16_t* p = Bbase + (size_t)(16 * j + row) * K + k0 + half * 16;
      b[j] = *(const v16bf*)p;
    }
#pragma unroll
    for (int i = 0; i < 4; ++i)
#pragma unroll
      for (int j = 0; j < 4; ++j)
        acc[i][j] = __builtin_amdgcn_wmma_f32_16x16x32_bf16(
            false, a[i], false, b[j], (short)0, acc[i][j], false, false);
  }

  // Epilogue: f32 C/D layout -> VGPR v holds row (half*8 + v), lane%16 holds column.
#pragma unroll
  for (int i = 0; i < 4; ++i) {
    const int rbase = M0 + 16 * i + half * 8;
#pragma unroll
    for (int j = 0; j < 4; ++j) {
      const int c = N0 + 16 * j + row;
      if (OUT_F32) {
        float* o = (float*)Cout;
        const float sc = s1[c];
        const float bs = bias[c];
#pragma unroll
        for (int v = 0; v < 8; ++v)
          o[(size_t)(rbase + v) * N + c] = acc[i][j][v] * sc + bs;
      } else {
        bf16_t* o = (bf16_t*)Cout;
#pragma unroll
        for (int v = 0; v < 8; ++v)
          o[(size_t)(rbase + v) * N + c] = (bf16_t)acc[i][j][v];
      }
    }
  }
}

extern "C" void kernel_launch(void* const* d_in, const int* in_sizes, int n_in,
                              void* d_out, int out_size, void* d_ws, size_t ws_size,
                              hipStream_t stream) {
  const float* x    = (const float*)d_in[0];
  const float* U    = (const float*)d_in[1];
  const float* V    = (const float*)d_in[2];
  const float* s1   = (const float*)d_in[3];
  const float* s2   = (const float*)d_in[4];
  const float* bias = (const float*)d_in[5];
  float* out = (float*)d_out;

  char* ws = (char*)d_ws;
  const size_t xs_bytes = (size_t)MTOK * DIN * sizeof(bf16_t);  // 64 MB
  const size_t us_bytes = (size_t)DOUT * RR * sizeof(bf16_t);   // 4 MB
  const size_t vs_bytes = (size_t)RR * DIN * sizeof(bf16_t);    // 4 MB
  bf16_t* xs  = (bf16_t*)(ws);
  bf16_t* usg = (bf16_t*)(ws + xs_bytes);
  bf16_t* vsT = (bf16_t*)(ws + xs_bytes + us_bytes);
  bf16_t* zz  = (bf16_t*)(ws + xs_bytes + us_bytes + vs_bytes); // 8 MB

  // Prep: activation scale+convert, binarize factors (V transposed to [R, DIN]).
  scale_x_kernel<<<(MTOK * (long long)DIN) / (8 * 256), 256, 0, stream>>>(x, s2, xs);
  sign_u_kernel<<<((long long)DOUT * RR) / (8 * 256), 256, 0, stream>>>(U, usg);
  sign_vT_kernel<<<dim3(RR / 32, DIN / 32), dim3(32, 8), 0, stream>>>(V, vsT);

  // GEMM1: z[8192,512] = xs[8192,4096] @ Vsign[4096,512]   (B stored as [512,4096])
  gemm_wmma_bf16<false><<<dim3(MTOK / 128, RR / 256), 256, 0, stream>>>(
      xs, vsT, zz, MTOK, RR, DIN, nullptr, nullptr);

  // GEMM2: out[8192,4096] = z[8192,512] @ Usign^T, fused *s1 + bias
  gemm_wmma_bf16<true><<<dim3(MTOK / 128, DOUT / 256), 256, 0, stream>>>(
      zz, usg, out, MTOK, DOUT, RR, s1, bias);
}
